// SortGRUEncoder_5617817223316
// MI455X (gfx1250) — compile-verified, matrix operand order checked by reading, not run on hardware
//
#include <hip/hip_runtime.h>
#include <math.h>

// ---------------------------------------------------------------------------
// SortGRUEncoder for MI455X (gfx1250).  Persistent-kernel design:
//   - all per-step GEMMs use v_wmma_f32_16x16x32_bf16 (bf16 A/B, f32 acc)
//   - weights converted once to bf16 (L2-resident), activations f32 master
//   - fragment loads vectorized as b128 (layout = two contiguous 8-elem runs)
//   - device-wide phases ordered by a global sense/generation barrier
//   - channel shuffles fused into WMMA A-fragment addressing
//   - per-row sort: bitonic in LDS, fused with GRU pointwise gates
// ---------------------------------------------------------------------------

#define DEVINL __device__ __forceinline__

typedef __bf16 bf16_t;
typedef bf16_t bfvec16 __attribute__((ext_vector_type(16)));
typedef bf16_t bfvec8  __attribute__((ext_vector_type(8)));
typedef float  fvec8   __attribute__((ext_vector_type(8)));
typedef float  fvec4   __attribute__((ext_vector_type(4)));

#define B_    64
#define T_    512
#define H_    512
#define L_SEQ 4096
#define C_    64
#define G3H   1536   // 3*H
#define NWG   32
#define NTHR  256

// ---- workspace layout (bytes) ----
static const size_t OFF_BAR = 0;                              // 256 B barrier
static const size_t OFF_WIH = 256;                            // 2*1536*512 bf16
static const size_t OFF_WHH = OFF_WIH + 2ull*1536*512*2;
static const size_t OFF_A1  = OFF_WHH + 2ull*1536*512*2;      // 2*512*512 bf16
static const size_t OFF_A2  = OFF_A1  + 2ull*512*512*2;       // 2*256*256 bf16
static const size_t OFF_A3  = OFF_A2  + 2ull*256*256*2;
static const size_t OFF_A4  = OFF_A3  + 2ull*256*256*2;       // 2*512*1024 bf16
static const size_t OFF_H   = OFF_A4  + 2ull*512*1024*2;      // 2*64*512 f32
static const size_t OFF_GI  = OFF_H   + 2ull*64*512*4;        // 64*1536 f32
static const size_t OFF_GH  = OFF_GI  + 64ull*1536*4;
static const size_t OFF_SE  = OFF_GH  + 64ull*1536*4;         // 128*512 f32
static const size_t OFF_YA  = OFF_SE  + 128ull*512*4;         // 64*1024 f32
static const size_t OFF_YB  = OFF_YA  + 64ull*1024*4;
static const size_t WS_NEED = OFF_YB  + 64ull*1024*4;

enum { EPI_NONE = 0, EPI_RELU = 1, EPI_SIGMUL = 2 };

DEVINL float sigf(float x) { return 1.0f / (1.0f + __expf(-x)); }

// channel-shuffle permutation over a 1024-vector viewed as [G=4][GH=256]
DEVINL int perm_idx(int i) { return ((i & 3) << 8) | (i >> 2); }

// ---- fragment loaders -----------------------------------------------------
// 16-bit WMMA fragment layout per lane: VGPR0..3 hold k = hi+0..7 (hi = 8 for
// lanes 16..31), VGPR4..7 hold k = 16+hi+0..7  ->  two contiguous 8-elem runs.

// A-fragment from f32 source, contiguous rows: 4x b128 loads + cvt_pk_bf16
DEVINL bfvec16 load_A_frag_vec(const float* rowbase, int k0, int lane) {
  const int hi = (lane & 16) ? 8 : 0;
  const float* p = rowbase + k0 + hi;
  fvec4 a0 = *(const fvec4*)(p);
  fvec4 a1 = *(const fvec4*)(p + 4);
  fvec4 a2 = *(const fvec4*)(p + 16);
  fvec4 a3 = *(const fvec4*)(p + 20);
  bfvec16 f;
#pragma unroll
  for (int i = 0; i < 4; ++i) {
    f[i]      = (bf16_t)a0[i];
    f[4 + i]  = (bf16_t)a1[i];
    f[8 + i]  = (bf16_t)a2[i];
    f[12 + i] = (bf16_t)a3[i];
  }
  return f;
}

// A-fragment with fused channel-shuffle gather (scattered -> scalar loads)
DEVINL bfvec16 load_A_frag_shuf(const float* base1024, int off, int k0, int lane) {
  const int hi = (lane & 16) ? 8 : 0;
  bfvec16 f;
#pragma unroll
  for (int i = 0; i < 8; ++i) {
    f[i]     = (bf16_t)base1024[perm_idx(off + k0 + hi + i)];
    f[8 + i] = (bf16_t)base1024[perm_idx(off + k0 + 16 + hi + i)];
  }
  return f;
}

// B-fragment from bf16 weight row: 2x b128 loads
DEVINL bfvec16 load_B_frag_vec(const bf16_t* wrow, int k0, int lane) {
  const int hi = (lane & 16) ? 8 : 0;
  bfvec8 lo = *(const bfvec8*)(wrow + k0 + hi);
  bfvec8 hi8 = *(const bfvec8*)(wrow + k0 + 16 + hi);
  bfvec16 f;
#pragma unroll
  for (int i = 0; i < 8; ++i) { f[i] = lo[i]; f[8 + i] = hi8[i]; }
  return f;
}

// ---- generic tiled GEMM ---------------------------------------------------
// C[M x N] = act( A[M x K](f32->bf16) * W[N x K]^T + bias )
// AMODE==1: A rows are (b,g) pairs over a [64][1024] buffer, fused shuffle.
template <int AMODE, int EPI, bool HASBIAS, bool HASOUT2>
DEVINL void gemm_tiles(int wid, int nwaves, int lane,
                       int M, int N, int K,
                       const float* A, long lda,
                       const bf16_t* W,
                       const float* bias,
                       float* C, int ldc,
                       const float* hmul, int ldh,
                       float* out2, long ldo) {
  const int nt = N >> 4;
  const int ntiles = (M >> 4) * nt;
  for (int tile = wid; tile < ntiles; tile += nwaves) {
    const int tm = tile / nt, tn = tile % nt;
    const int m0 = tm << 4, n0 = tn << 4;
    const int mrow = m0 + (lane & 15);
    const float* rowA;
    int off = 0;
    if (AMODE) { rowA = A + (long)(mrow >> 2) * lda; off = (mrow & 3) << 8; }
    else       { rowA = A + (long)mrow * lda; }
    const bf16_t* rowW = W + (long)(n0 + (lane & 15)) * K;

    fvec8 acc;
    const float binit = HASBIAS ? bias[n0 + (lane & 15)] : 0.0f;
#pragma unroll
    for (int r = 0; r < 8; ++r) acc[r] = binit;

    for (int k0 = 0; k0 < K; k0 += 32) {
      bfvec16 fa = AMODE ? load_A_frag_shuf(rowA, off, k0, lane)
                         : load_A_frag_vec(rowA, k0, lane);
      bfvec16 fb = load_B_frag_vec(rowW, k0, lane);
      acc = __builtin_amdgcn_wmma_f32_16x16x32_bf16(
          false, fa, false, fb, (short)0, acc, false, false);
    }

    const int col = n0 + (lane & 15);
#pragma unroll
    for (int r = 0; r < 8; ++r) {
      const int row = m0 + r + ((lane & 16) ? 8 : 0);
      float v = acc[r];
      if (EPI == EPI_RELU)   v = v > 0.0f ? v : 0.0f;
      if (EPI == EPI_SIGMUL) v = sigf(v) * hmul[(long)row * ldh + col];
      C[(long)row * ldc + col] = v;
      if (HASOUT2) out2[(long)row * ldo + col] = v;
    }
  }
}

// device-wide barrier: generation counter in scratch, s_sleep spin
DEVINL void grid_barrier(unsigned* bar) {
  __syncthreads();
  if (threadIdx.x == 0) {
    __threadfence();
    unsigned gen = __hip_atomic_load(&bar[1], __ATOMIC_RELAXED, __HIP_MEMORY_SCOPE_AGENT);
    unsigned arrived = __hip_atomic_fetch_add(&bar[0], 1u, __ATOMIC_ACQ_REL,
                                              __HIP_MEMORY_SCOPE_AGENT);
    if (arrived == gridDim.x - 1) {
      __hip_atomic_store(&bar[0], 0u, __ATOMIC_RELAXED, __HIP_MEMORY_SCOPE_AGENT);
      __hip_atomic_store(&bar[1], gen + 1u, __ATOMIC_RELEASE, __HIP_MEMORY_SCOPE_AGENT);
    } else {
      while (__hip_atomic_load(&bar[1], __ATOMIC_ACQUIRE, __HIP_MEMORY_SCOPE_AGENT) == gen)
        __builtin_amdgcn_s_sleep(1);
    }
    __threadfence();
  }
  __syncthreads();
}

DEVINL void bitonic_sort_512(float* d) {
  const int t = threadIdx.x;   // 256 threads, 512 elements
  for (int k = 2; k <= 512; k <<= 1) {
    for (int j = k >> 1; j > 0; j >>= 1) {
      __syncthreads();
      if (t < 256) {
        const int i = ((t & ~(j - 1)) << 1) | (t & (j - 1));
        const int p = i | j;
        const float a = d[i], b = d[p];
        const bool up = ((i & k) == 0);
        if ((a > b) == up) { d[i] = b; d[p] = a; }
      }
    }
  }
  __syncthreads();
}

// ---------------------------------------------------------------------------
__global__ void sortgru_init(const float* __restrict__ hidden0,
                             const float* __restrict__ w_ih,
                             const float* __restrict__ w_hh,
                             const float* __restrict__ a1,
                             const float* __restrict__ a2,
                             const float* __restrict__ a3,
                             const float* __restrict__ a4,
                             unsigned char* __restrict__ ws) {
  const long tid  = (long)blockIdx.x * blockDim.x + threadIdx.x;
  const long nthr = (long)gridDim.x * blockDim.x;
  if (tid < 64) ((unsigned*)(ws + OFF_BAR))[tid] = 0u;

  bf16_t* wih = (bf16_t*)(ws + OFF_WIH);
  bf16_t* whh = (bf16_t*)(ws + OFF_WHH);
  bf16_t* b1  = (bf16_t*)(ws + OFF_A1);
  bf16_t* b2  = (bf16_t*)(ws + OFF_A2);
  bf16_t* b3  = (bf16_t*)(ws + OFF_A3);
  bf16_t* b4  = (bf16_t*)(ws + OFF_A4);
  float*  hb  = (float*)(ws + OFF_H);

  for (long i = tid; i < 2l * G3H * H_; i += nthr) wih[i] = (bf16_t)w_ih[i];
  for (long i = tid; i < 2l * G3H * H_; i += nthr) whh[i] = (bf16_t)w_hh[i];
  for (long i = tid; i < 2l * H_ * H_;  i += nthr) b1[i]  = (bf16_t)a1[i];
  for (long i = tid; i < 2l * 256 * 256; i += nthr) b2[i] = (bf16_t)a2[i];
  for (long i = tid; i < 2l * 256 * 256; i += nthr) b3[i] = (bf16_t)a3[i];
  for (long i = tid; i < 2l * H_ * 1024; i += nthr) b4[i] = (bf16_t)a4[i];
  for (long i = tid; i < 2l * B_ * H_; i += nthr) {
    const int l = (int)(i / (B_ * H_));
    const int n = (int)(i & (H_ - 1));
    hb[i] = hidden0[l * H_ + n];
  }
}

// ---------------------------------------------------------------------------
__global__ void __launch_bounds__(NTHR)
sortgru_main(const float* __restrict__ x,
             const float* __restrict__ b_ih,
             const float* __restrict__ b_hh,
             float* __restrict__ out,
             unsigned char* __restrict__ ws) {
  __shared__ float s_sort[512];

  unsigned* bar = (unsigned*)(ws + OFF_BAR);
  const bf16_t* wih = (const bf16_t*)(ws + OFF_WIH);
  const bf16_t* whh = (const bf16_t*)(ws + OFF_WHH);
  const bf16_t* a1b = (const bf16_t*)(ws + OFF_A1);
  const bf16_t* a2b = (const bf16_t*)(ws + OFF_A2);
  const bf16_t* a3b = (const bf16_t*)(ws + OFF_A3);
  const bf16_t* a4b = (const bf16_t*)(ws + OFF_A4);
  float* hbuf = (float*)(ws + OFF_H);   // [2][64][512]
  float* gi   = (float*)(ws + OFF_GI);  // [64][1536]
  float* gh   = (float*)(ws + OFF_GH);  // [64][1536]
  float* se   = (float*)(ws + OFF_SE);  // [128][512]  (h,sorted interleaved)
  float* yA   = (float*)(ws + OFF_YA);  // [64][1024]
  float* yB   = (float*)(ws + OFF_YB);  // [64][1024]

  const int lane   = threadIdx.x & 31;
  const int wid    = blockIdx.x * (blockDim.x >> 5) + (threadIdx.x >> 5);
  const int nwaves = gridDim.x * (blockDim.x >> 5);

  for (int t = 0; t < T_; ++t) {
    for (int l = 0; l < 2; ++l) {
      float* hl = hbuf + (long)l * B_ * H_;
      const float* Agi = l ? hbuf : (x + (long)t * 512);
      const long lda_gi = l ? H_ : (long)L_SEQ * C_;

      // ---- phase 1: gate GEMMs  gi = xt*Wih^T+bih, gh = h*Whh^T+bhh ----
      gemm_tiles<0, EPI_NONE, true, false>(wid, nwaves, lane, B_, G3H, H_,
          Agi, lda_gi, wih + (long)l * G3H * H_, b_ih + l * G3H,
          gi, G3H, nullptr, 0, nullptr, 0);
      gemm_tiles<0, EPI_NONE, true, false>(wid, nwaves, lane, B_, G3H, H_,
          hl, H_, whh + (long)l * G3H * H_, b_hh + l * G3H,
          gh, G3H, nullptr, 0, nullptr, 0);
      grid_barrier(bar);

      // ---- phase 2: GRU pointwise + per-row bitonic sort -> se ----
      for (int row = blockIdx.x; row < B_; row += gridDim.x) {
        const float* gib = gi + (long)row * G3H;
        const float* ghb = gh + (long)row * G3H;
        float* hrow = hl + (long)row * H_;
        for (int e = threadIdx.x; e < H_; e += blockDim.x) {
          const float r = sigf(gib[e] + ghb[e]);
          const float z = sigf(gib[H_ + e] + ghb[H_ + e]);
          const float n = tanhf(gib[2 * H_ + e] + r * ghb[2 * H_ + e]);
          const float hv = (1.0f - z) * n + z * hrow[e];
          hrow[e] = hv;
          se[(long)(2 * row) * H_ + e] = hv;
          s_sort[e] = hv;
        }
        __syncthreads();
        bitonic_sort_512(s_sort);
        for (int e = threadIdx.x; e < H_; e += blockDim.x)
          se[(long)(2 * row + 1) * H_ + e] = s_sort[e];
        __syncthreads();
      }
      grid_barrier(bar);

      // ---- phase 3: y = se @ a1^T    [128x512]x[512x512] -> yA ----
      gemm_tiles<0, EPI_NONE, false, false>(wid, nwaves, lane, 128, H_, H_,
          se, H_, a1b + (long)l * H_ * H_, nullptr,
          yA, H_, nullptr, 0, nullptr, 0);
      grid_barrier(bar);

      // ---- phase 4: shuffle + group GEMM a2  [256x256]x[256x256] -> yB ----
      gemm_tiles<1, EPI_NONE, false, false>(wid, nwaves, lane, 256, 256, 256,
          yA, 1024, a2b + (long)l * 256 * 256, nullptr,
          yB, 256, nullptr, 0, nullptr, 0);
      grid_barrier(bar);

      // ---- phase 5: shuffle + group GEMM a3 + ReLU -> yA ----
      gemm_tiles<1, EPI_RELU, false, false>(wid, nwaves, lane, 256, 256, 256,
          yB, 1024, a3b + (long)l * 256 * 256, nullptr,
          yA, 256, nullptr, 0, nullptr, 0);
      grid_barrier(bar);

      // ---- phase 6: scores=sigmoid(y@a4^T); h = h*scores (+output) ----
      if (l == 1) {
        gemm_tiles<0, EPI_SIGMUL, false, true>(wid, nwaves, lane, B_, H_, 1024,
            yA, 1024, a4b + (long)l * H_ * 1024, nullptr,
            hl, H_, hl, H_, out + (long)t * H_, (long)T_ * H_);
      } else {
        gemm_tiles<0, EPI_SIGMUL, false, false>(wid, nwaves, lane, B_, H_, 1024,
            yA, 1024, a4b + (long)l * H_ * 1024, nullptr,
            hl, H_, hl, H_, nullptr, 0);
      }
      grid_barrier(bar);
    }
  }
}

// ---------------------------------------------------------------------------
extern "C" void kernel_launch(void* const* d_in, const int* in_sizes, int n_in,
                              void* d_out, int out_size, void* d_ws, size_t ws_size,
                              hipStream_t stream) {
  (void)in_sizes; (void)n_in; (void)out_size;
  if (ws_size < WS_NEED) return;

  const float* x       = (const float*)d_in[0];
  const float* hidden0 = (const float*)d_in[1];
  const float* w_ih    = (const float*)d_in[2];
  const float* w_hh    = (const float*)d_in[3];
  const float* b_ih    = (const float*)d_in[4];
  const float* b_hh    = (const float*)d_in[5];
  const float* a1      = (const float*)d_in[6];
  const float* a2      = (const float*)d_in[7];
  const float* a3      = (const float*)d_in[8];
  const float* a4      = (const float*)d_in[9];
  float* out = (float*)d_out;
  unsigned char* ws = (unsigned char*)d_ws;

  sortgru_init<<<dim3(1024), dim3(NTHR), 0, stream>>>(hidden0, w_ih, w_hh,
                                                      a1, a2, a3, a4, ws);
  sortgru_main<<<dim3(NWG), dim3(NTHR), 0, stream>>>(x, b_ih, b_hh, out, ws);
}
